// Attention_42451456753780
// MI455X (gfx1250) — matrix operand order repaired and checked by reading, not compiled, not run on hardware
//
#include <hip/hip_runtime.h>
#include <hip/hip_bf16.h>

// ---------------------------------------------------------------------------
// Problem constants (from reference): B=2, S=2048, D=1024, H=16, HD=64
// ---------------------------------------------------------------------------
#define BB 2
#define SS 2048
#define DD 1024
#define HH 16
#define HD 64
#define ROWS (BB * SS)          // 4096
#define NEG_INF (-3.0e38f)

typedef __attribute__((ext_vector_type(16))) __bf16 v16bf;
typedef __attribute__((ext_vector_type(8)))  float  v8f;
typedef __attribute__((ext_vector_type(4)))  unsigned int u32x4;
typedef __attribute__((ext_vector_type(8)))  int i32x8;
typedef __attribute__((ext_vector_type(4)))  int i32x4;

union Frag {                    // 16 bf16 = 8 dwords per lane
    v16bf v;
    unsigned int u[8];
};

__device__ __forceinline__ unsigned short f2bf(float f) {
    unsigned int u = __float_as_uint(f);
    unsigned int r = u + 0x7fffu + ((u >> 16) & 1u);   // round-to-nearest-even
    return (unsigned short)(r >> 16);
}

__device__ __forceinline__ v8f wmma_bf16(const Frag& a, const Frag& b, v8f c) {
    return __builtin_amdgcn_wmma_f32_16x16x32_bf16(
        /*neg_a=*/false, a.v, /*neg_b=*/false, b.v,
        /*c_mod=*/(short)0, c, /*reuse_a=*/false, /*reuse_b=*/false);
}

// A-fragment (16x32 bf16, M x K): lane m=lane&15, half=lane>>4,
// dword v holds K = (v>>2)*16 + half*8 + (v&3)*2. 32-bit indexing off the base.
__device__ __forceinline__ void loadA(Frag& a, const unsigned short* base,
                                      int row0, int ld, int col0, int lane) {
    const int m = lane & 15, half = lane >> 4;
    const int idx = (row0 + m) * ld + col0 + half * 8;
#pragma unroll
    for (int g = 0; g < 2; ++g) {
#pragma unroll
        for (int j = 0; j < 4; ++j)
            a.u[g * 4 + j] = *(const unsigned int*)(base + idx + g * 16 + 2 * j);
    }
}

// B-fragment (32x16 bf16, K x N): lane n=lane&15, half=lane>>4,
// dword v holds K = (v>>2)*16 + half*8 + (v&3)*2. Source is "n-major, K contiguous".
__device__ __forceinline__ void loadB(Frag& b, const unsigned short* base,
                                      int n0, int ldn, int k0, int lane) {
    const int n = lane & 15, half = lane >> 4;
    const int idx = (n0 + n) * ldn + k0 + half * 8;
#pragma unroll
    for (int v = 0; v < 8; ++v)
        b.u[v] = *(const unsigned int*)(base + idx + (v >> 2) * 16 + (v & 3) * 2);
}

// ---------------------------------------------------------------------------
// Tensor Data Mover: 2D tile load global -> LDS (D# per CDNA5 ISA ch.8).
// data_size = 2 bytes; groups 2/3 zero (2D tile, tile_dim2 = 0).
// ---------------------------------------------------------------------------
#define HAS_TDM __has_builtin(__builtin_amdgcn_tensor_load_to_lds)

#if HAS_TDM
__device__ __forceinline__ void tdm_load_2d(unsigned lds_byte_off,
                                            const unsigned short* gptr,
                                            unsigned tensor_d0, unsigned tensor_d1,
                                            unsigned tile_d0, unsigned tile_d1,
                                            unsigned stride0 /*elements*/) {
    const unsigned long long ga = (unsigned long long)gptr;
    u32x4 g0 = {
        1u,                                               // count=1 (user D#)
        lds_byte_off,                                     // lds_addr
        (unsigned)(ga & 0xffffffffu),                     // global_addr[31:0]
        (unsigned)((ga >> 32) & 0x1ffffffu) | (2u << 30)  // addr[56:32] | type=2
    };
    i32x8 g1 = {
        (int)(1u << 16),                                          // data_size=1 -> 2B
        (int)((tensor_d0 & 0xffffu) << 16),                       // tdim0[15:0]
        (int)((tensor_d0 >> 16) | ((tensor_d1 & 0xffffu) << 16)), // tdim0[31:16]|tdim1[15:0]
        (int)((tensor_d1 >> 16) | (tile_d0 << 16)),               // tdim1[31:16]|tile_d0
        (int)tile_d1,                                             // tile_d1 (tile_d2=0)
        (int)stride0,                                             // dim0 stride low32
        0, 0
    };
    i32x4 z4 = {0, 0, 0, 0};
#if defined(__clang_major__) && (__clang_major__ >= 23)
    i32x8 z8 = {0, 0, 0, 0, 0, 0, 0, 0};
    __builtin_amdgcn_tensor_load_to_lds(g0, g1, z4, z4, z8, 0);
#else
    __builtin_amdgcn_tensor_load_to_lds(g0, g1, z4, z4, 0);
#endif
}
#endif

// ---------------------------------------------------------------------------
// 0) fp32 -> bf16 conversion
// ---------------------------------------------------------------------------
__global__ void cvt_bf16_kernel(const float* __restrict__ in,
                                unsigned short* __restrict__ out, int n) {
    int i = blockIdx.x * blockDim.x + threadIdx.x;
    const int stride = gridDim.x * blockDim.x;
    for (; i < n; i += stride) out[i] = f2bf(in[i]);
}

// ---------------------------------------------------------------------------
// 1) QKV projection + RoPE, one matrix per blockIdx.z (0=Q,1=K,2=V).
//    One wave: 32 rows x 64 cols (one head). grid = (ROWS/32, H, 3), block = 32.
//    Outputs: q,k as [B*H][S][64] bf16 ; v transposed as [B*H][64][S] bf16.
// ---------------------------------------------------------------------------
__global__ void proj_kernel(const unsigned short* __restrict__ xb,
                            const unsigned short* __restrict__ wq,
                            const unsigned short* __restrict__ wk,
                            const unsigned short* __restrict__ wv,
                            unsigned short* __restrict__ qo,
                            unsigned short* __restrict__ ko,
                            unsigned short* __restrict__ vTo) {
    const int lane = threadIdx.x;
    const int row0 = blockIdx.x * 32;
    const int h = blockIdx.y;
    const int z = blockIdx.z;
    const unsigned short* __restrict__ w = (z == 0) ? wq : (z == 1) ? wk : wv;
    const int nn = lane & 15, halfsel = lane >> 4;

    v8f acc[2][4] = {};

    for (int c0 = 0; c0 < DD; c0 += 32) {
        Frag a0, a1, b0, b1;
        loadA(a0, xb, row0,      DD, c0, lane);
        loadA(a1, xb, row0 + 16, DD, c0, lane);
#pragma unroll
        for (int t = 0; t < 4; t += 2) {
            loadB(b0, w, h * HD + t * 16,       DD, c0, lane);
            loadB(b1, w, h * HD + (t + 1) * 16, DD, c0, lane);
            acc[0][t]     = wmma_bf16(a0, b0, acc[0][t]);
            acc[1][t]     = wmma_bf16(a1, b0, acc[1][t]);
            acc[0][t + 1] = wmma_bf16(a0, b1, acc[0][t + 1]);
            acc[1][t + 1] = wmma_bf16(a1, b1, acc[1][t + 1]);
        }
    }

    // RoPE for Q and K: pairs (d, d+32) are tiles t and t+2 at identical slots.
    if (z < 2) {
#pragma unroll
        for (int t = 0; t < 2; ++t) {
            const int d1 = t * 16 + nn;                   // 0..31
            const float invf = (d1 < 16) ? __powf(1024.0f, -(float)d1 / 15.0f) : 0.0f;
#pragma unroll
            for (int mt = 0; mt < 2; ++mt) {
#pragma unroll
                for (int r = 0; r < 8; ++r) {
                    const int rowg = row0 + mt * 16 + r + halfsel * 8;
                    const int s = rowg & (SS - 1);
                    const float ang = (float)s * invf;
                    const float sn = __sinf(ang), cs = __cosf(ang);
                    const float t1 = acc[mt][t][r], t2 = acc[mt][t + 2][r];
                    acc[mt][t][r]     = t1 * cs - t2 * sn;
                    acc[mt][t + 2][r] = t1 * sn + t2 * cs;
                }
            }
        }
    }

#pragma unroll
    for (int mt = 0; mt < 2; ++mt) {
#pragma unroll
        for (int t = 0; t < 4; ++t) {
            const int d = t * 16 + nn;
#pragma unroll
            for (int r = 0; r < 8; ++r) {
                const int rowg = row0 + mt * 16 + r + halfsel * 8;
                const int bb = rowg >> 11;                // / SS
                const int s = rowg & (SS - 1);
                const int bh = bb * HH + h;
                const unsigned short val = f2bf(acc[mt][t][r]);
                if (z == 0)      qo[(bh * SS + s) * HD + d]  = val;
                else if (z == 1) ko[(bh * SS + s) * HD + d]  = val;
                else             vTo[(bh * HD + d) * SS + s] = val;
            }
        }
    }
}

// ---------------------------------------------------------------------------
// 2) Causal flash attention.
//    Workgroup = 4 waves = 64 q rows of one (b,h); K/V slabs for each 32-key
//    block staged into LDS once per workgroup by the Tensor Data Mover and
//    shared by all 4 waves. grid = (S/64, B*H), block = 128.
//    LDS: K slab [32][64] bf16 | V slab [64][32] bf16 | 4x P tile [16][32] bf16.
// ---------------------------------------------------------------------------
#define SK 0
#define SV 2048
#define SP 4096

__global__ void attn_kernel(const unsigned short* __restrict__ q,
                            const unsigned short* __restrict__ kmat,
                            const unsigned short* __restrict__ vT,
                            unsigned short* __restrict__ ctx) {
    __shared__ __align__(16) unsigned short smem[SP + 4 * 512];   // 12 KB

    const int lane = threadIdx.x & 31;
    const int wid = threadIdx.x >> 5;
    const int q0blk = blockIdx.x * 64;
    const int q0 = q0blk + wid * 16;
    const int bh = blockIdx.y;
    const int nn = lane & 15, halfsel = lane >> 4;

    const unsigned short* __restrict__ qb = q    + bh * SS * HD;
    const unsigned short* __restrict__ kb = kmat + bh * SS * HD;
    const unsigned short* __restrict__ vb = vT   + bh * HD * SS;

    const unsigned short* kslab = &smem[SK];
    const unsigned short* vslab = &smem[SV];
    unsigned short* pld = &smem[SP + wid * 512];

    Frag qa0, qa1;
    loadA(qa0, qb, q0, HD, 0,  lane);
    loadA(qa1, qb, q0, HD, 32, lane);

    v8f o0 = {}, o1 = {}, o2 = {}, o3 = {};
    float m8[8], l8[8];
#pragma unroll
    for (int r = 0; r < 8; ++r) { m8[r] = NEG_INF; l8[r] = 0.0f; }

    const float scale = 0.125f;                           // 1/sqrt(64)
    const int myTmax = (q0 + 15) >> 5;                    // this wave's causal bound
    const int blkTmax = (q0blk + 63) >> 5;                // block-wide bound

#if HAS_TDM
    const unsigned ldsbase = (unsigned)(unsigned long long)(const void*)&smem[0];
#endif

    for (int t = 0; t <= blkTmax; ++t) {
        const int k0 = t * 32;

#if HAS_TDM
        if (threadIdx.x < 32) {                           // wave 0 drives the TDM
            // K slab: rows k0..k0+31 of k[bh] (row len 64) -> LDS [32][64]
            tdm_load_2d(ldsbase + SK * 2, kb + k0 * HD,
                        HD, (unsigned)(SS - k0), HD, 32, HD);
            // V slab: cols k0..k0+31 over 64 rows of vT[bh] -> LDS [64][32]
            tdm_load_2d(ldsbase + SV * 2, vb + k0,
                        (unsigned)(SS - k0), HD, 32, HD, SS);
            __builtin_amdgcn_s_wait_tensorcnt(0);
        }
        __syncthreads();                                  // publish slabs
#else
        {   // cooperative fallback copy
            const unsigned short* gk = kb + k0 * HD;
            const unsigned short* gv = vb + k0;
            for (int i = threadIdx.x; i < 256; i += 128)  // 32*64 shorts / 8
                *(uint4*)&smem[SK + i * 8] = *(const uint4*)&gk[i * 8];
            for (int i = threadIdx.x; i < 256; i += 128) {
                const int d = i >> 2, c = i & 3;
                *(uint4*)&smem[SV + d * 32 + c * 8] = *(const uint4*)&gv[d * SS + c * 8];
            }
        }
        __syncthreads();
#endif

        if (t <= myTmax) {
            v8f s0 = {}, s1 = {};
            Frag b0, b1;
            // scores: Q(16x64) x K^T(64x32) from LDS K slab [key][d]
            loadB(b0, kslab, 0,  HD, 0,  lane);
            loadB(b1, kslab, 0,  HD, 32, lane);
            s0 = wmma_bf16(qa0, b0, s0);
            s0 = wmma_bf16(qa1, b1, s0);
            loadB(b0, kslab, 16, HD, 0,  lane);
            loadB(b1, kslab, 16, HD, 32, lane);
            s1 = wmma_bf16(qa0, b0, s1);
            s1 = wmma_bf16(qa1, b1, s1);

            const bool needmask = (k0 + 31 > q0);
#pragma unroll
            for (int r = 0; r < 8; ++r) {
                float x0 = s0[r] * scale, x1 = s1[r] * scale;
                if (needmask) {
                    const int qrow = q0 + r + halfsel * 8;
                    if (k0 + nn      > qrow) x0 = NEG_INF;
                    if (k0 + 16 + nn > qrow) x1 = NEG_INF;
                }
                s0[r] = x0; s1[r] = x1;
            }

            // online softmax: per-row (VGPR r) reduce across the 16-lane half
#pragma unroll
            for (int r = 0; r < 8; ++r) {
                float mx = fmaxf(s0[r], s1[r]);
#pragma unroll
                for (int off = 1; off < 16; off <<= 1)
                    mx = fmaxf(mx, __shfl_xor(mx, off, 32));
                const float mnew = fmaxf(m8[r], mx);
                const float alpha = __expf(m8[r] - mnew);
                const float p0 = __expf(s0[r] - mnew);
                const float p1 = __expf(s1[r] - mnew);
                float rs = p0 + p1;
#pragma unroll
                for (int off = 1; off < 16; off <<= 1)
                    rs += __shfl_xor(rs, off, 32);
                l8[r] = l8[r] * alpha + rs;
                m8[r] = mnew;
                s0[r] = p0; s1[r] = p1;
                o0[r] *= alpha; o1[r] *= alpha; o2[r] *= alpha; o3[r] *= alpha;
            }

            // C-layout -> A-layout re-staging of P (per-wave LDS region;
            // same-wave DS ops are in order, no barrier needed)
#pragma unroll
            for (int r = 0; r < 8; ++r) {
                const int mrow = r + halfsel * 8;
                pld[mrow * 32 + nn]      = f2bf(s0[r]);
                pld[mrow * 32 + 16 + nn] = f2bf(s1[r]);
            }
            Frag pa;
            {
                const int m = lane & 15, half = lane >> 4;
#pragma unroll
                for (int v = 0; v < 8; ++v) {
                    const int kk = (v >> 2) * 16 + half * 8 + (v & 3) * 2;
                    pa.u[v] = *(const unsigned int*)&pld[m * 32 + kk];
                }
            }

            // O += P(16x32) x V(32x64) from LDS V slab [d][key]
            loadB(b0, vslab, 0,  32, 0, lane);
            loadB(b1, vslab, 16, 32, 0, lane);
            o0 = wmma_bf16(pa, b0, o0);
            o1 = wmma_bf16(pa, b1, o1);
            loadB(b0, vslab, 32, 32, 0, lane);
            loadB(b1, vslab, 48, 32, 0, lane);
            o2 = wmma_bf16(pa, b0, o2);
            o3 = wmma_bf16(pa, b1, o3);
        }
        __syncthreads();                                  // WAR before next slab
    }

    const int h = bh & (HH - 1);
    const int bb = bh >> 4;
#pragma unroll
    for (int r = 0; r < 8; ++r) {
        const float inv = 1.0f / l8[r];
        const int s = q0 + r + halfsel * 8;
        const int rowg = bb * SS + s;
        unsigned short* dst = ctx + rowg * DD + h * HD + nn;
        dst[0]  = f2bf(o0[r] * inv);
        dst[16] = f2bf(o1[r] * inv);
        dst[32] = f2bf(o2[r] * inv);
        dst[48] = f2bf(o3[r] * inv);
    }
}

// ---------------------------------------------------------------------------
// 3) Output projection: out = ctx @ Wo^T + bo (fp32 out).
//    One wave: 32 rows x 64 cols. grid = (ROWS/32, D/64), block = 32.
// ---------------------------------------------------------------------------
__global__ void oproj_kernel(const unsigned short* __restrict__ ctx,
                             const unsigned short* __restrict__ wob,
                             const float* __restrict__ bo,
                             float* __restrict__ out) {
    const int lane = threadIdx.x;
    const int row0 = blockIdx.x * 32;
    const int j0 = blockIdx.y * 64;
    const int nn = lane & 15, halfsel = lane >> 4;

    v8f acc[2][4] = {};
    for (int c0 = 0; c0 < DD; c0 += 32) {
        Frag a0, a1, b0, b1;
        loadA(a0, ctx, row0,      DD, c0, lane);
        loadA(a1, ctx, row0 + 16, DD, c0, lane);
#pragma unroll
        for (int t = 0; t < 4; t += 2) {
            loadB(b0, wob, j0 + t * 16,       DD, c0, lane);
            loadB(b1, wob, j0 + (t + 1) * 16, DD, c0, lane);
            acc[0][t]     = wmma_bf16(a0, b0, acc[0][t]);
            acc[1][t]     = wmma_bf16(a1, b0, acc[1][t]);
            acc[0][t + 1] = wmma_bf16(a0, b1, acc[0][t + 1]);
            acc[1][t + 1] = wmma_bf16(a1, b1, acc[1][t + 1]);
        }
    }
#pragma unroll
    for (int t = 0; t < 4; ++t) {
        const int j = j0 + t * 16 + nn;
        const float bias = bo[j];
#pragma unroll
        for (int mt = 0; mt < 2; ++mt) {
#pragma unroll
            for (int r = 0; r < 8; ++r) {
                const int rowg = row0 + mt * 16 + r + halfsel * 8;
                out[rowg * DD + j] = acc[mt][t][r] + bias;
            }
        }
    }
}

// ---------------------------------------------------------------------------
// Host-side launcher
// ---------------------------------------------------------------------------
extern "C" void kernel_launch(void* const* d_in, const int* in_sizes, int n_in,
                              void* d_out, int out_size, void* d_ws, size_t ws_size,
                              hipStream_t stream) {
    const float* x  = (const float*)d_in[0];
    const float* Wq = (const float*)d_in[1];
    const float* Wk = (const float*)d_in[2];
    const float* Wv = (const float*)d_in[3];
    const float* Wo = (const float*)d_in[4];
    const float* bo = (const float*)d_in[5];
    // d_in[6] = mask (causal, handled analytically)

    const size_t nX = (size_t)ROWS * DD;           // 4,194,304
    const size_t nW = (size_t)DD * DD;             // 1,048,576
    const size_t nQKV = (size_t)BB * HH * SS * HD; // 4,194,304

    char* ws = (char*)d_ws;
    size_t off = 0;
    unsigned short* xb  = (unsigned short*)(ws + off); off += nX * 2;
    unsigned short* wqb = (unsigned short*)(ws + off); off += nW * 2;
    unsigned short* wkb = (unsigned short*)(ws + off); off += nW * 2;
    unsigned short* wvb = (unsigned short*)(ws + off); off += nW * 2;
    unsigned short* wob = (unsigned short*)(ws + off); off += nW * 2;
    unsigned short* qh  = (unsigned short*)(ws + off); off += nQKV * 2;
    unsigned short* kh  = (unsigned short*)(ws + off); off += nQKV * 2;
    unsigned short* vTh = (unsigned short*)(ws + off); off += nQKV * 2;
    unsigned short* ctx = (unsigned short*)(ws + off); off += nX * 2;

    // 0) convert inputs to bf16
    cvt_bf16_kernel<<<4096, 256, 0, stream>>>(x,  xb,  (int)nX);
    cvt_bf16_kernel<<<2048, 256, 0, stream>>>(Wq, wqb, (int)nW);
    cvt_bf16_kernel<<<2048, 256, 0, stream>>>(Wk, wkb, (int)nW);
    cvt_bf16_kernel<<<2048, 256, 0, stream>>>(Wv, wvb, (int)nW);
    cvt_bf16_kernel<<<2048, 256, 0, stream>>>(Wo, wob, (int)nW);

    // 1) QKV projection + RoPE (z: 0=Q, 1=K, 2=V)
    proj_kernel<<<dim3(ROWS / 32, HH, 3), 32, 0, stream>>>(
        xb, wqb, wkb, wvb, qh, kh, vTh);

    // 2) causal flash attention (TDM-staged K/V slabs, 4 waves/WG)
    attn_kernel<<<dim3(SS / 64, BB * HH), 128, 0, stream>>>(qh, kh, vTh, ctx);

    // 3) output projection + bias
    oproj_kernel<<<dim3(ROWS / 32, DD / 64), 32, 0, stream>>>(
        ctx, wob, bo, (float*)d_out);
}
